// RSSMDynamics_70093866271164
// MI455X (gfx1250) — compile-verified, matrix operand order checked by reading, not run on hardware
//
#include <hip/hip_runtime.h>
#include <hip/hip_bf16.h>
#include <math.h>

// ---------------------------------------------------------------------------
// RSSM dynamics on MI455X (gfx1250, wave32, WMMA).
// Design: batch-parallel persistent workgroups. B=256 split into 16 blocks of
// 16 rows (WMMA M dim); each block's whole T=256 scan runs inside one
// workgroup (32 wave32s) with activations resident in ~261KB of LDS.
// Weights converted once to f16 in d_ws, resident in L2 (7MB of 192MB), and
// streamed per-step as WMMA B-fragments via global_load_b128. A laundered
// 32-bit OFFSET (not pointer!) defeats LICM so the loads are neither
// hoisted+spilled to scratch nor demoted to flat_load. The k-loop is
// software-pipelined (depth 1) to overlap L2 latency with v_wmma.
// All GEMMs use v_wmma_f32_16x16x32_f16 (f32 accumulate).
// ---------------------------------------------------------------------------

#define D_  256
#define H_  512
#define S_  32
#define T_  256
#define B_  256
#define BB  16          // batch rows per workgroup (= WMMA M)
#define NBLK (B_ / BB)  // 16 workgroups
#define G3H (3 * H_)    // 1536

typedef __attribute__((ext_vector_type(16))) _Float16 v16h;
typedef __attribute__((ext_vector_type(8)))  _Float16 v8h;
typedef __attribute__((ext_vector_type(4)))  _Float16 v4h;
typedef __attribute__((ext_vector_type(8)))  float    v8f;
typedef __attribute__((ext_vector_type(4)))  float    v4f;

// f16 weight offsets inside d_ws (elements)
#define OFF_WIH0    0u
#define OFF_WHH0    442368u           // + 1536*288
#define OFF_WIH1    1228800u          // + 1536*512
#define OFF_WHH1    2015232u
#define OFF_WPOST1  2801664u
#define OFF_WPOST2  3194880u          // + 512*768
#define OFF_WPRIOR1 3227648u          // + 64*512
#define OFF_WPRIOR2 3489792u
#define WS_F16_ELEMS 3522560u         // ~7.05 MB

// output layout (f32): stoch, prm, prlv, pm, plv  [B,T,S] each, then h_final [L,B,H]
#define BTS (B_ * T_ * S_)
#define O_STOCH 0
#define O_PRM   (1 * BTS)
#define O_PRLV  (2 * BTS)
#define O_PM    (3 * BTS)
#define O_PLV   (4 * BTS)
#define O_HF    (5 * BTS)

// LDS carve (bytes)
#define L_XIN    0          // f16 [16][768]  = [h1 | x_t]
#define L_RNNIN  24576      // f16 [16][288]  = [x_t | stoch]
#define L_H0H    33792      // f16 [16][512]
#define L_PRH    50176      // f16 [16][512]  hidden of post/prior MLP
#define L_G      66560      // f32 [16][1536] gate pre-activations
#define L_GHN    164864     // f32 [16][512]  hn part (kept separate)
#define L_H0F    197632     // f32 [16][512]
#define L_H1F    230400     // f32 [16][512]
#define L_PP     263168     // f32 [16][64]
#define LDS_BYTES 267264

// ---------------------------------------------------------------------------

__global__ void cvt_f32_f16(const float* __restrict__ src,
                            _Float16* __restrict__ dst, int n) {
  int i = blockIdx.x * blockDim.x + threadIdx.x;
  if (i < n) dst[i] = (_Float16)src[i];
}

// Load a 16-lane WMMA fragment: elems 0..7 = p[0..7], elems 8..15 = p[16..23]
// (per ISA 7.12.2 16-bit A/B layout; caller folds lane>>4 into p).
__device__ __forceinline__ v16h ld_frag(const _Float16* p) {
  v8h lo = *(const v8h*)(p);
  v8h hi = *(const v8h*)(p + 16);
  v16h r;
#pragma unroll
  for (int i = 0; i < 8; ++i) { r[i] = lo[i]; r[i + 8] = hi[i]; }
  return r;
}

// acc += A_tile(rows 0..15, K) x W_tile(rows n0..n0+15, K)^T
// A row-major [m][k] (LDS), W row-major [n][k] (global f16, L2-resident).
// Depth-1 software pipeline: loads for k+32 are issued before the WMMA on k.
__device__ __forceinline__ v8f gemm_tile(const _Float16* __restrict__ A, int lda,
                                         const _Float16* __restrict__ W, int ldw,
                                         int n0, int K, int lane, v8f acc) {
  const int r    = lane & 15;
  const int koff = (lane >> 4) << 3;
  const _Float16* ap = A + r * lda + koff;
  const _Float16* wp = W + (size_t)(n0 + r) * ldw + koff;
  v16h a = ld_frag(ap);
  v16h b = ld_frag(wp);
  for (int k = 32; k < K; k += 32) {
    v16h an = ld_frag(ap + k);
    v16h bn = ld_frag(wp + k);
    acc = __builtin_amdgcn_wmma_f32_16x16x32_f16(
        false, a, false, b, (short)0, acc, false, false);
    a = an; b = bn;
  }
  return __builtin_amdgcn_wmma_f32_16x16x32_f16(
      false, a, false, b, (short)0, acc, false, false);
}

__device__ __forceinline__ float sigm(float x) {
  return 1.0f / (1.0f + __expf(-x));
}

// ---------------------------------------------------------------------------

__global__ void __launch_bounds__(1024, 1)
rssm_kernel(const float* __restrict__ latent,   // [B,T,D]
            const float* __restrict__ hidden,   // [L,B,H]
            const float* __restrict__ eps,      // [B,T,S]
            const _Float16* __restrict__ wsh,   // converted f16 weights
            const float* __restrict__ bih0, const float* __restrict__ bhh0,
            const float* __restrict__ bih1, const float* __restrict__ bhh1,
            const float* __restrict__ bpost1, const float* __restrict__ bpost2,
            const float* __restrict__ bprior1, const float* __restrict__ bprior2,
            float* __restrict__ out) {
  extern __shared__ char smem[];
  _Float16* xin   = (_Float16*)(smem + L_XIN);
  _Float16* rnnin = (_Float16*)(smem + L_RNNIN);
  _Float16* h0h   = (_Float16*)(smem + L_H0H);
  _Float16* prh   = (_Float16*)(smem + L_PRH);
  float*    g     = (float*)(smem + L_G);
  float*    ghn   = (float*)(smem + L_GHN);
  float*    h0f   = (float*)(smem + L_H0F);
  float*    h1f   = (float*)(smem + L_H1F);
  float*    pp    = (float*)(smem + L_PP);

  const int tid  = threadIdx.x;
  const int lane = tid & 31;
  const int wave = tid >> 5;         // 0..31
  const int lrow = lane & 15;
  const int half = lane >> 4;
  const int gb0  = blockIdx.x * BB;  // first batch row of this block

  // ---- init hidden state from input (h=0 per reference, but honor buffer)
  for (int idx = tid; idx < BB * H_; idx += 1024) {
    int m = idx >> 9, j = idx & (H_ - 1);
    float v0 = hidden[(size_t)(0 * B_ + gb0 + m) * H_ + j];
    float v1 = hidden[(size_t)(1 * B_ + gb0 + m) * H_ + j];
    h0f[m * H_ + j] = v0;  h0h[m * H_ + j] = (_Float16)v0;
    h1f[m * H_ + j] = v1;  xin[m * 768 + j] = (_Float16)v1;   // xin[:,0:512] = h1
  }
  __syncthreads();

  for (int t = 0; t < T_; ++t) {
    // Launder a 32-bit OFFSET (value 0) through opaque asm: the weight
    // addresses become loop-variant (no LICM hoist -> no scratch spill),
    // but pointer provenance stays with the kernel argument, so the loads
    // keep the GLOBAL address space (global_load_b128, not flat_load).
    unsigned woff = 0;
    asm volatile("" : "+s"(woff));
    const _Float16* wih0h    = wsh + OFF_WIH0    + woff;
    const _Float16* whh0h    = wsh + OFF_WHH0    + woff;
    const _Float16* wih1h    = wsh + OFF_WIH1    + woff;
    const _Float16* whh1h    = wsh + OFF_WHH1    + woff;
    const _Float16* wpost1h  = wsh + OFF_WPOST1  + woff;
    const _Float16* wpost2h  = wsh + OFF_WPOST2  + woff;
    const _Float16* wprior1h = wsh + OFF_WPRIOR1 + woff;
    const _Float16* wprior2h = wsh + OFF_WPRIOR2 + woff;

    // ---- Phase A: stage x_t into xin[:,512:768] and rnnin[:,0:256] (f16)
    {
      int m  = tid >> 6;            // 0..15
      int k4 = (tid & 63) << 2;     // 0..252
      const float* lp = latent + ((size_t)(gb0 + m) * T_ + t) * D_ + k4;
      v4f x = *(const v4f*)lp;
      v4h xh;
#pragma unroll
      for (int i = 0; i < 4; ++i) xh[i] = (_Float16)x[i];
      *(v4h*)(&xin[m * 768 + 512 + k4]) = xh;   // 8B-aligned ds_store_b64
      *(v4h*)(&rnnin[m * 288 + k4])     = xh;   // 8B-aligned ds_store_b64
    }
    __syncthreads();

    // ---- Phase B: posterior layer 1: prh = relu(xin @ wpost1^T + b), 32 tiles
    {
      int n0 = wave * 16;
      v8f acc = {};
      acc = gemm_tile(xin, 768, wpost1h, 768, n0, 768, lane, acc);
      float b = bpost1[n0 + lrow];
#pragma unroll
      for (int i = 0; i < 8; ++i) {
        float v = acc[i] + b;
        v = v > 0.0f ? v : 0.0f;
        prh[(half * 8 + i) * H_ + n0 + lrow] = (_Float16)v;
      }
    }
    __syncthreads();

    // ---- Phase C: posterior layer 2: pp = prh @ wpost2^T + b  (4 tiles)
    if (wave < 4) {
      int n0 = wave * 16;
      v8f acc = {};
      acc = gemm_tile(prh, H_, wpost2h, H_, n0, H_, lane, acc);
      float b = bpost2[n0 + lrow];
#pragma unroll
      for (int i = 0; i < 8; ++i)
        pp[(half * 8 + i) * 64 + n0 + lrow] = acc[i] + b;
    }
    __syncthreads();

    // ---- Phase D: reparam sample; emit stoch/pm/plv; fill rnnin[:,256:288]
    if (tid < BB * S_) {
      int m = tid >> 5, s = tid & (S_ - 1);
      float pm  = pp[m * 64 + s];
      float plv = pp[m * 64 + 32 + s];
      float e   = eps[((size_t)(gb0 + m) * T_ + t) * S_ + s];
      float st  = pm + e * __expf(0.5f * plv);
      size_t o = ((size_t)(gb0 + m) * T_ + t) * S_ + s;
      out[O_STOCH + o] = st;
      out[O_PM + o]    = pm;
      out[O_PLV + o]   = plv;
      rnnin[m * 288 + 256 + s] = (_Float16)st;
    }
    __syncthreads();

    // ---- Phase E: GRU0 gates: 96 N-tiles over 32 waves (tt = r/z/n third)
#pragma unroll
    for (int tt = 0; tt < 3; ++tt) {
      int n0 = (wave + tt * 32) * 16;
      v8f acc = {};
      acc = gemm_tile(rnnin, 288, wih0h, 288, n0, 288, lane, acc);  // x part
      if (tt < 2) {   // r,z: gi+gh summed
        acc = gemm_tile(h0h, H_, whh0h, H_, n0, H_, lane, acc);
        float b = bih0[n0 + lrow] + bhh0[n0 + lrow];
#pragma unroll
        for (int i = 0; i < 8; ++i)
          g[(half * 8 + i) * G3H + n0 + lrow] = acc[i] + b;
      } else {        // n: keep inn and hn separate
        v8f acch = {};
        acch = gemm_tile(h0h, H_, whh0h, H_, n0, H_, lane, acch);
        float bi = bih0[n0 + lrow], bh = bhh0[n0 + lrow];
#pragma unroll
        for (int i = 0; i < 8; ++i) {
          g[(half * 8 + i) * G3H + n0 + lrow]            = acc[i] + bi;
          ghn[(half * 8 + i) * H_ + (n0 - 1024) + lrow]  = acch[i] + bh;
        }
      }
    }
    __syncthreads();

    // ---- Phase F: GRU0 elementwise update -> h0f, h0h
    for (int idx = tid; idx < BB * H_; idx += 1024) {
      int m = idx >> 9, j = idx & (H_ - 1);
      float r = sigm(g[m * G3H + j]);
      float z = sigm(g[m * G3H + H_ + j]);
      float n = tanhf(g[m * G3H + 2 * H_ + j] + r * ghn[m * H_ + j]);
      float hn = (1.0f - z) * n + z * h0f[m * H_ + j];
      h0f[m * H_ + j] = hn;
      h0h[m * H_ + j] = (_Float16)hn;
    }
    __syncthreads();

    // ---- Phase G: GRU1 gates (A_i = h0h, A_h = old h1 still in xin[:,0:512])
#pragma unroll
    for (int tt = 0; tt < 3; ++tt) {
      int n0 = (wave + tt * 32) * 16;
      v8f acc = {};
      acc = gemm_tile(h0h, H_, wih1h, H_, n0, H_, lane, acc);
      if (tt < 2) {
        acc = gemm_tile(xin, 768, whh1h, H_, n0, H_, lane, acc);
        float b = bih1[n0 + lrow] + bhh1[n0 + lrow];
#pragma unroll
        for (int i = 0; i < 8; ++i)
          g[(half * 8 + i) * G3H + n0 + lrow] = acc[i] + b;
      } else {
        v8f acch = {};
        acch = gemm_tile(xin, 768, whh1h, H_, n0, H_, lane, acch);
        float bi = bih1[n0 + lrow], bh = bhh1[n0 + lrow];
#pragma unroll
        for (int i = 0; i < 8; ++i) {
          g[(half * 8 + i) * G3H + n0 + lrow]           = acc[i] + bi;
          ghn[(half * 8 + i) * H_ + (n0 - 1024) + lrow] = acch[i] + bh;
        }
      }
    }
    __syncthreads();

    // ---- Phase H: GRU1 elementwise -> h1f (f32), xin[:,0:512] (f16, new h1)
    for (int idx = tid; idx < BB * H_; idx += 1024) {
      int m = idx >> 9, j = idx & (H_ - 1);
      float r = sigm(g[m * G3H + j]);
      float z = sigm(g[m * G3H + H_ + j]);
      float n = tanhf(g[m * G3H + 2 * H_ + j] + r * ghn[m * H_ + j]);
      float hn = (1.0f - z) * n + z * h1f[m * H_ + j];
      h1f[m * H_ + j] = hn;
      xin[m * 768 + j] = (_Float16)hn;
    }
    __syncthreads();

    // ---- Phase I: prior layer 1: prh = relu(h1 @ wprior1^T + b)
    {
      int n0 = wave * 16;
      v8f acc = {};
      acc = gemm_tile(xin, 768, wprior1h, H_, n0, H_, lane, acc);
      float b = bprior1[n0 + lrow];
#pragma unroll
      for (int i = 0; i < 8; ++i) {
        float v = acc[i] + b;
        v = v > 0.0f ? v : 0.0f;
        prh[(half * 8 + i) * H_ + n0 + lrow] = (_Float16)v;
      }
    }
    __syncthreads();

    // ---- Phase J: prior layer 2 -> pp
    if (wave < 4) {
      int n0 = wave * 16;
      v8f acc = {};
      acc = gemm_tile(prh, H_, wprior2h, H_, n0, H_, lane, acc);
      float b = bprior2[n0 + lrow];
#pragma unroll
      for (int i = 0; i < 8; ++i)
        pp[(half * 8 + i) * 64 + n0 + lrow] = acc[i] + b;
    }
    __syncthreads();

    // ---- Phase K: emit prm/prlv
    {
      int m = tid >> 6, s2 = tid & 63;
      float v = pp[m * 64 + s2];
      size_t o = ((size_t)(gb0 + m) * T_ + t) * S_;
      if (s2 < 32) out[O_PRM + o + s2] = v;
      else         out[O_PRLV + o + (s2 - 32)] = v;
    }
    __syncthreads();
  }

  // ---- h_final [L,B,H]
  for (int idx = tid; idx < BB * H_; idx += 1024) {
    int m = idx >> 9, j = idx & (H_ - 1);
    out[O_HF + (size_t)(0 * B_ + gb0 + m) * H_ + j] = h0f[m * H_ + j];
    out[O_HF + (size_t)(1 * B_ + gb0 + m) * H_ + j] = h1f[m * H_ + j];
  }
}

// ---------------------------------------------------------------------------

extern "C" void kernel_launch(void* const* d_in, const int* in_sizes, int n_in,
                              void* d_out, int out_size, void* d_ws, size_t ws_size,
                              hipStream_t stream) {
  const float* latent  = (const float*)d_in[0];
  const float* hidden  = (const float*)d_in[1];
  const float* eps     = (const float*)d_in[2];
  const float* wih0    = (const float*)d_in[3];
  const float* whh0    = (const float*)d_in[4];
  const float* bih0    = (const float*)d_in[5];
  const float* bhh0    = (const float*)d_in[6];
  const float* wih1    = (const float*)d_in[7];
  const float* whh1    = (const float*)d_in[8];
  const float* bih1    = (const float*)d_in[9];
  const float* bhh1    = (const float*)d_in[10];
  const float* wpost1  = (const float*)d_in[11];
  const float* bpost1  = (const float*)d_in[12];
  const float* wpost2  = (const float*)d_in[13];
  const float* bpost2  = (const float*)d_in[14];
  const float* wprior1 = (const float*)d_in[15];
  const float* bprior1 = (const float*)d_in[16];
  const float* wprior2 = (const float*)d_in[17];
  const float* bprior2 = (const float*)d_in[18];

  _Float16* wsh = (_Float16*)d_ws;
  auto cvt = [&](const float* s, unsigned off, int n) {
    cvt_f32_f16<<<(n + 255) / 256, 256, 0, stream>>>(s, wsh + off, n);
  };
  cvt(wih0,    OFF_WIH0,    G3H * (D_ + S_));
  cvt(whh0,    OFF_WHH0,    G3H * H_);
  cvt(wih1,    OFF_WIH1,    G3H * H_);
  cvt(whh1,    OFF_WHH1,    G3H * H_);
  cvt(wpost1,  OFF_WPOST1,  H_ * (H_ + D_));
  cvt(wpost2,  OFF_WPOST2,  2 * S_ * H_);
  cvt(wprior1, OFF_WPRIOR1, H_ * H_);
  cvt(wprior2, OFF_WPRIOR2, 2 * S_ * H_);

  (void)hipFuncSetAttribute(reinterpret_cast<const void*>(rssm_kernel),
                            hipFuncAttributeMaxDynamicSharedMemorySize,
                            LDS_BYTES);

  rssm_kernel<<<dim3(NBLK), dim3(1024), LDS_BYTES, stream>>>(
      latent, hidden, eps, wsh,
      bih0, bhh0, bih1, bhh1, bpost1, bpost2, bprior1, bprior2,
      (float*)d_out);
}
